// GatingBiTTTLayer_8761733284544
// MI455X (gfx1250) — matrix operand order repaired and checked by reading, not compiled, not run on hardware
//
#include <hip/hip_runtime.h>

#define MROWS 32768
#define CDIM 384
#define NPATCH 256
#define KLEN 128
#define NHEADS 6
#define HDF 64
#define MC (MROWS*CDIM)

typedef __bf16 bf16_t;
typedef __attribute__((ext_vector_type(16))) __bf16 v16bf;
typedef __attribute__((ext_vector_type(8)))  float  v8f;

__device__ __forceinline__ bf16_t tobf(float x){
  unsigned u = __builtin_bit_cast(unsigned, x);
  unsigned r = u + 0x7fffu + ((u >> 16) & 1u);   // round-to-nearest-even
  unsigned short h = (unsigned short)(r >> 16);
  return __builtin_bit_cast(bf16_t, h);
}

__device__ __forceinline__ v8f vzero8(){
  v8f z = {0.f,0.f,0.f,0.f,0.f,0.f,0.f,0.f};
  return z;
}

__device__ __forceinline__ v8f wmma_bf16(v16bf a, v16bf b, v8f c){
  return __builtin_amdgcn_wmma_f32_16x16x32_bf16(false, a, false, b, (short)0, c, false, false);
}

// A fragment 16x32 bf16, from row-major LDS tile (rows = M, cols = K), leading dim ld.
// ISA layout: lane L: M = L%16; K in {kb..kb+7} U {kb+16..kb+23}, kb = (L<16)?0:8.
__device__ __forceinline__ v16bf load_afrag(const bf16_t* s, int ld, int row0, int kk){
  int lane = threadIdx.x & 31;
  const bf16_t* p = s + (row0 + (lane & 15)) * ld + kk + ((lane >> 4) << 3);
  v16bf a;
#pragma unroll
  for (int e = 0; e < 8; ++e){ a[e] = p[e]; a[e+8] = p[16+e]; }
  return a;
}

// B fragment 32x16 bf16, from n-major LDS tile (rows = N, cols = K), leading dim ld.
// ISA layout: lane L: N = L%16; K in {kb..kb+7} U {kb+16..kb+23}, kb = (L<16)?0:8.
__device__ __forceinline__ v16bf load_bfrag(const bf16_t* s, int ld, int n0, int kk){
  int lane = threadIdx.x & 31;
  const bf16_t* p = s + (n0 + (lane & 15)) * ld + kk + ((lane >> 4) << 3);
  v16bf b;
#pragma unroll
  for (int e = 0; e < 8; ++e){ b[e] = p[e]; b[e+8] = p[16+e]; }
  return b;
}

// ---------------- Generic bf16 GEMM: C[M,384] = A[M,384] @ W[384,384] (+bias) ----------------
// block 256 = 8 waves (4 M x 2 N), block tile 128x128, K chunk 32.
__global__ __launch_bounds__(256) void gemm_bf16_384(
    const bf16_t* __restrict__ A, const bf16_t* __restrict__ W,
    const float* __restrict__ bias, float* __restrict__ C)
{
  __shared__ bf16_t As[128*40];
  __shared__ bf16_t Bs[128*40];
  int m0 = blockIdx.x * 128;
  int n0 = blockIdx.y * 128;
  int tid = threadIdx.x;
  int w  = tid >> 5;
  int wm = (w & 3) << 5;   // 32-row wave tile
  int wn = (w >> 2) << 6;  // 64-col wave tile
  v8f acc[2][4];
#pragma unroll
  for (int i=0;i<2;++i)
#pragma unroll
    for (int j=0;j<4;++j) acc[i][j]=vzero8();

  for (int kk = 0; kk < CDIM; kk += 32){
    { // stage A tile 128x32 (row-major, M x K)
      int r = tid >> 1, c = (tid & 1) << 4;
      const bf16_t* src = A + (size_t)(m0 + r) * CDIM + kk + c;
      bf16_t* dst = As + r*40 + c;
#pragma unroll
      for (int e=0;e<16;++e) dst[e]=src[e];
    }
    { // stage B tile as [n][k] (n-major so frag loads are contiguous)
      int n = tid & 127, k0 = (tid >> 7) << 4;
      const bf16_t* src = W + (size_t)(kk + k0) * CDIM + n0 + n;
      bf16_t* dst = Bs + n*40 + k0;
#pragma unroll
      for (int e=0;e<16;++e) dst[e]=src[(size_t)e*CDIM];
    }
    __syncthreads();
    v16bf a0 = load_afrag(As, 40, wm,    0);
    v16bf a1 = load_afrag(As, 40, wm+16, 0);
#pragma unroll
    for (int j=0;j<4;++j){
      v16bf bb = load_bfrag(Bs, 40, wn + (j<<4), 0);
      acc[0][j] = wmma_bf16(a0, bb, acc[0][j]);
      acc[1][j] = wmma_bf16(a1, bb, acc[1][j]);
    }
    __syncthreads();
  }
  int lane = tid & 31;
#pragma unroll
  for (int i=0;i<2;++i){
#pragma unroll
    for (int j=0;j<4;++j){
      int n  = n0 + wn + (j<<4) + (lane & 15);
      int mb = m0 + wm + (i<<4) + ((lane >> 4) << 3);
      float bv = bias ? bias[n] : 0.f;
#pragma unroll
      for (int r=0;r<8;++r)
        C[(size_t)(mb + r) * CDIM + n] = acc[i][j][r] + bv;
    }
  }
}

// ---------------- TTT inner per (patch b, head h): dual-form mini-batch ----------------
__global__ __launch_bounds__(256) void ttt_inner(
    const bf16_t* __restrict__ hq16, const bf16_t* __restrict__ hk16,
    const float* __restrict__ hqf,   const float* __restrict__ tgt,
    const float* __restrict__ lre,   const bf16_t* __restrict__ W1t,
    const float* __restrict__ b1p,   const float* __restrict__ lnw,
    const float* __restrict__ lnbp,  const float* __restrict__ tokidx,
    float* __restrict__ outc)
{
  __shared__ bf16_t sQ[128*72];     // XQ [m][f]
  __shared__ bf16_t sK[128*72];     // XK [m][f]
  __shared__ bf16_t sW[64*72];      // W1 as [g][f]   (B layout: N=g, K=f)
  __shared__ bf16_t sA[128*136];    // -mask*eta*(Attn+1)  [m][l] (A layout)
  __shared__ bf16_t sG[64*136];     // grad^T [f][l]  (B layout: N=f, K=l)
  __shared__ float  sZ[128*68];     // f32 staging: Z1, then xhat, then Z1_bar
  __shared__ float  sLre[128], sTok[128], sB1[64], sLw[64], sLb[64];

  int h = blockIdx.x, b = blockIdx.y;
  int tid = threadIdx.x;
  size_t base = ((size_t)(b * NHEADS + h)) * (KLEN * HDF);

  for (int i = tid; i < KLEN*HDF; i += 256){
    int r = i >> 6, c = i & 63;
    sQ[r*72+c] = hq16[base + i];
    sK[r*72+c] = hk16[base + i];
  }
  for (int i = tid; i < HDF*HDF; i += 256){
    int r = i >> 6, c = i & 63;
    sW[r*72+c] = W1t[(size_t)h*4096 + i];
  }
  if (tid < 128){
    sLre[tid] = lre[((size_t)(b*NHEADS+h))*KLEN + tid];
    float tv = 1.f/(float)(tid+1) + tokidx[tid];
    sTok[tid] = tv > 0.f ? tv : 0.f;
  }
  if (tid < 64){
    sB1[tid] = b1p[h*64+tid];
    sLw[tid] = lnw[h*64+tid];
    sLb[tid] = lnbp[h*64+tid];
  }
  __syncthreads();

  int w = tid >> 5;
  int m0 = w << 4;       // 16 rows per wave
  int lane = tid & 31;

  // Z1 = XK @ W1 + b1
  {
    v8f z[4];
#pragma unroll
    for (int j=0;j<4;++j) z[j]=vzero8();
    for (int kk=0; kk<64; kk+=32){
      v16bf a = load_afrag(sK, 72, m0, kk);
#pragma unroll
      for (int j=0;j<4;++j){
        v16bf bb = load_bfrag(sW, 72, j<<4, kk);
        z[j] = wmma_bf16(a, bb, z[j]);
      }
    }
#pragma unroll
    for (int j=0;j<4;++j){
      int n  = (j<<4) + (lane & 15);
      int mb = m0 + ((lane>>4)<<3);
#pragma unroll
      for (int r=0;r<8;++r) sZ[(mb+r)*68+n] = z[j][r] + sB1[n];
    }
  }
  __syncthreads();

  // grad = LN-L2 backward vs (XV - XK); one row per thread (threads 0..127)
  if (tid < 128){
    int m = tid;
    float mu = 0.f;
#pragma unroll
    for (int c=0;c<64;++c) mu += sZ[m*68+c];
    mu *= (1.f/64.f);
    float var = 0.f;
#pragma unroll
    for (int c=0;c<64;++c){ float d = sZ[m*68+c]-mu; var += d*d; }
    var *= (1.f/64.f);
    float rstd = rsqrtf(var + 1e-6f);
    float gx[64];
    float mg=0.f, mgx=0.f;
#pragma unroll
    for (int c=0;c<64;++c){
      float xh = (sZ[m*68+c]-mu)*rstd;
      sZ[m*68+c] = xh;
      float g = (sLw[c]*xh + sLb[c] - tgt[base + m*64 + c]) * sLw[c];
      gx[c]=g; mg+=g; mgx+=g*xh;
    }
    mg *= (1.f/64.f); mgx *= (1.f/64.f);
#pragma unroll
    for (int c=0;c<64;++c){
      float gr = (gx[c] - mg - sZ[m*68+c]*mgx) * rstd;
      sG[c*136+m] = tobf(gr);           // transposed: [f][l]
    }
  }

  // Attn = XQ @ XK^T ; build  -mask*eta*(Attn+1)  (folds b1_bar's mask*eta term)
  {
    v8f at[8];
#pragma unroll
    for (int j=0;j<8;++j) at[j]=vzero8();
    for (int kk=0; kk<64; kk+=32){
      v16bf a = load_afrag(sQ, 72, m0, kk);
#pragma unroll
      for (int j=0;j<8;++j){
        v16bf bb = load_bfrag(sK, 72, j<<4, kk);
        at[j] = wmma_bf16(a, bb, at[j]);
      }
    }
#pragma unroll
    for (int j=0;j<8;++j){
      int l = (j<<4) + (lane & 15);
      float le = sLre[l];
      int mb = m0 + ((lane>>4)<<3);
#pragma unroll
      for (int r=0;r<8;++r){
        int m = mb + r;
        float v = (l <= m) ? (-sTok[m] * le * (at[j][r] + 1.f)) : 0.f;
        sA[m*136+l] = tobf(v);
      }
    }
  }
  __syncthreads();

  // Z1_bar = XQ@W1 + b1 - [mask*eta*(Attn+1)] @ grad
  {
    v8f zb[4];
#pragma unroll
    for (int j=0;j<4;++j) zb[j]=vzero8();
    for (int kk=0;kk<64;kk+=32){
      v16bf a = load_afrag(sQ, 72, m0, kk);
#pragma unroll
      for (int j=0;j<4;++j){
        v16bf bb = load_bfrag(sW, 72, j<<4, kk);
        zb[j] = wmma_bf16(a, bb, zb[j]);
      }
    }
    for (int kk=0;kk<128;kk+=32){
      v16bf a = load_afrag(sA, 136, m0, kk);
#pragma unroll
      for (int j=0;j<4;++j){
        v16bf bb = load_bfrag(sG, 136, j<<4, kk);
        zb[j] = wmma_bf16(a, bb, zb[j]);
      }
    }
#pragma unroll
    for (int j=0;j<4;++j){
      int n  = (j<<4) + (lane & 15);
      int mb = m0 + ((lane>>4)<<3);
#pragma unroll
      for (int r=0;r<8;++r) sZ[(mb+r)*68+n] = zb[j][r] + sB1[n];
    }
  }
  __syncthreads();

  // out = XQ + LN(Z1_bar); write to [B,K,C] with c = h*64+f
  if (tid < 128){
    int m = tid;
    float mu=0.f;
#pragma unroll
    for (int c=0;c<64;++c) mu += sZ[m*68+c];
    mu *= (1.f/64.f);
    float var=0.f;
#pragma unroll
    for (int c=0;c<64;++c){ float d=sZ[m*68+c]-mu; var+=d*d; }
    var *= (1.f/64.f);
    float rstd = rsqrtf(var+1e-6f);
    size_t orow = (size_t)(b*KLEN + m) * CDIM + h*HDF;
#pragma unroll
    for (int c=0;c<64;++c){
      float xh = (sZ[m*68+c]-mu)*rstd;
      outc[orow + c] = hqf[base + m*64 + c] + sLw[c]*xh + sLb[c];
    }
  }
}

// ---------------- elementwise / conversion kernels ----------------
__global__ void k_convert_x(const float* __restrict__ x, bf16_t* __restrict__ xn, bf16_t* __restrict__ xf){
  int idx = blockIdx.x*256 + threadIdx.x;
  if (idx >= MC) return;
  int c = idx % CDIM; int row = idx / CDIM;
  int t = row & 127, b = row >> 7;
  xn[idx] = tobf(x[idx]);
  int rf = (b<<7) + (127 - t);
  xf[idx] = tobf(x[(size_t)rf*CDIM + c]);
}

__global__ void k_convert_plain(const float* __restrict__ s, bf16_t* __restrict__ d, int n){
  int idx = blockIdx.x*256 + threadIdx.x;
  if (idx < n) d[idx] = tobf(s[idx]);
}

__global__ void k_convert_w1t(const float* __restrict__ s, bf16_t* __restrict__ d){
  int idx = blockIdx.x*256 + threadIdx.x;
  if (idx >= NHEADS*64*64) return;
  int h = idx >> 12, rem = idx & 4095;
  int g = rem >> 6, f = rem & 63;
  d[idx] = tobf(s[h*4096 + f*64 + g]);   // [h][g][f] = W1[h][f][g]
}

// causal depthwise conv (k=4, left pad 3) + interleaved RoPE, into head layout
__global__ void k_conv_rope(const float* __restrict__ xq,
    const float* __restrict__ cqw, const float* __restrict__ cqb,
    const float* __restrict__ ckw, const float* __restrict__ ckb,
    bf16_t* __restrict__ hq16, bf16_t* __restrict__ hk16,
    float* __restrict__ hqf, float* __restrict__ hkf)
{
  int idx = blockIdx.x*256 + threadIdx.x;
  if (idx >= MROWS*(CDIM/2)) return;
  int cp = idx % (CDIM/2); int row = idx / (CDIM/2);
  int t = row & 127, b = row >> 7;
  int c0 = cp << 1;
  float q0=cqb[c0], q1=cqb[c0+1], k0=ckb[c0], k1=ckb[c0+1];
#pragma unroll
  for (int j=0;j<4;++j){
    int tt = t + j - 3;
    if (tt >= 0){
      size_t rr = (size_t)((b<<7)+tt)*CDIM;
      float v0 = xq[rr + c0], v1 = xq[rr + c0 + 1];
      q0 += cqw[c0*4+j]*v0; q1 += cqw[(c0+1)*4+j]*v1;
      k0 += ckw[c0*4+j]*v0; k1 += ckw[(c0+1)*4+j]*v1;
    }
  }
  int h  = c0 / HDF;
  int f0 = c0 % HDF;
  int i  = f0 >> 1;
  float inv = powf(10000.f, -((float)(2*i))/64.f);
  float ang = (float)t * inv;
  float cs = cosf(ang), sn = sinf(ang);
  float qe = q0*cs - q1*sn, qo = q1*cs + q0*sn;
  float ke = k0*cs - k1*sn, ko = k1*cs + k0*sn;
  size_t o = ((size_t)(b*NHEADS+h)*KLEN + t)*HDF + f0;
  hqf[o]=qe; hqf[o+1]=qo; hkf[o]=ke; hkf[o+1]=ko;
  hq16[o]=tobf(qe); hq16[o+1]=tobf(qo);
  hk16[o]=tobf(ke); hk16[o+1]=tobf(ko);
}

// tgt = XV - XK in [b,h,t,f]; in place over the XK f32 buffer
__global__ void k_make_tgt(const float* __restrict__ gv, float* __restrict__ tg){
  int idx = blockIdx.x*256 + threadIdx.x;
  if (idx >= MC) return;
  int f = idx & 63; int r = idx >> 6;
  int t = r & 127; int bh = r >> 7;
  int hh = bh % NHEADS, b = bh / NHEADS;
  size_t gi = (size_t)((b<<7)+t)*CDIM + hh*HDF + f;
  tg[idx] = gv[gi] - tg[idx];
}

__global__ void k_lr(const float* __restrict__ x, const float* __restrict__ lrw,
                     const float* __restrict__ lrb, float* __restrict__ lre, int flip){
  int idx = blockIdx.x*256 + threadIdx.x;
  if (idx >= NPATCH*NHEADS*KLEN) return;
  int t = idx & 127; int bh = idx >> 7;
  int h = bh % NHEADS, b = bh / NHEADS;
  int row = (b<<7) + (flip ? (127 - t) : t);
  float s = lrb[h];
  const float* xr = x + (size_t)row*CDIM;
  const float* wr = lrw + h*CDIM;
  for (int c=0;c<CDIM;++c) s += xr[c]*wr[c];
  lre[idx] = (1.f/(1.f+expf(-s))) * (1.f/64.f);
}

// z = LN( gelu_tanh(gg) * core ) -> bf16, one row per block of 128 threads
__global__ __launch_bounds__(128) void k_gate_postnorm(
    const float* __restrict__ gg, const float* __restrict__ core,
    const float* __restrict__ pnw, const float* __restrict__ pnb,
    bf16_t* __restrict__ o16)
{
  __shared__ float red[8];
  int r = blockIdx.x;
  int tid = threadIdx.x;
  float v[3];
  float s=0.f, s2=0.f;
#pragma unroll
  for (int e=0;e<3;++e){
    int c = tid + e*128;
    float g = gg[(size_t)r*CDIM + c];
    float ge = 0.5f*g*(1.f + tanhf(0.7978845608f*(g + 0.044715f*g*g*g)));
    float val = ge * core[(size_t)r*CDIM + c];
    v[e]=val; s+=val; s2+=val*val;
  }
#pragma unroll
  for (int off=16; off; off>>=1){ s += __shfl_xor(s, off); s2 += __shfl_xor(s2, off); }
  int w = tid >> 5;
  if ((tid & 31)==0){ red[w]=s; red[4+w]=s2; }
  __syncthreads();
  s  = red[0]+red[1]+red[2]+red[3];
  s2 = red[4]+red[5]+red[6]+red[7];
  float mu = s/384.f;
  float var = s2/384.f - mu*mu;
  float rstd = rsqrtf(var + 1e-6f);
#pragma unroll
  for (int e=0;e<3;++e){
    int c = tid + e*128;
    o16[(size_t)r*CDIM + c] = tobf(pnw[c]*((v[e]-mu)*rstd) + pnb[c]);
  }
}

__global__ void k_fuse(const float* __restrict__ of, const float* __restrict__ ob,
                       const float* __restrict__ x, const float* __restrict__ gf,
                       const float* __restrict__ gb, float* __restrict__ y){
  int idx = blockIdx.x*256 + threadIdx.x;
  if (idx >= MC) return;
  int c = idx % CDIM; int row = idx / CDIM;
  int t = row & 127, b = row >> 7;
  int rf = (b<<7) + (127-t);
  y[idx] = tanhf(gf[0])*of[idx] + tanhf(gb[0])*ob[(size_t)rf*CDIM+c] + x[idx];
}

__global__ void k_dwconv3_bf16(const float* __restrict__ in, const float* __restrict__ wv,
                               const float* __restrict__ bv, bf16_t* __restrict__ o16){
  int idx = blockIdx.x*256 + threadIdx.x;
  if (idx >= MC) return;
  int c = idx % CDIM; int row = idx / CDIM;
  int t = row & 127, b = row >> 7;
  float a = bv[c];
#pragma unroll
  for (int j=0;j<3;++j){
    int tt = t + j - 1;
    if (tt >= 0 && tt < 128) a += wv[c*3+j]*in[(size_t)((b<<7)+tt)*CDIM + c];
  }
  o16[idx] = tobf(a);
}

__global__ void k_silumul(const float* __restrict__ t1, const float* __restrict__ t2,
                          bf16_t* __restrict__ u16){
  int idx = blockIdx.x*256 + threadIdx.x;
  if (idx >= MC) return;
  float a = t1[idx];
  float s = a/(1.f+expf(-a));
  u16[idx] = tobf(s * t2[idx]);
}

__global__ void k_final(const float* __restrict__ y, const float* __restrict__ v,
                        const float* __restrict__ wv, const float* __restrict__ bv,
                        float* __restrict__ out){
  int idx = blockIdx.x*256 + threadIdx.x;
  if (idx >= MC) return;
  int c = idx % CDIM; int row = idx / CDIM;
  int t = row & 127, b = row >> 7;
  float a = bv[c];
#pragma unroll
  for (int j=0;j<3;++j){
    int tt = t + j - 1;
    if (tt >= 0 && tt < 128) a += wv[c*3+j]*v[(size_t)((b<<7)+tt)*CDIM + c];
  }
  out[idx] = y[idx] + a;
}

// ---------------- host orchestration ----------------
extern "C" void kernel_launch(void* const* d_in, const int* in_sizes, int n_in,
                              void* d_out, int out_size, void* d_ws, size_t ws_size,
                              hipStream_t stream)
{
  (void)in_sizes; (void)n_in; (void)out_size; (void)ws_size;
  // input index map (setup_inputs dict order, leaves flattened):
  // 0 data, 1 depth, 2..18 params_fwd {wq,wv,wo,wg,cq_w,cq_b,ck_w,ck_b,lr_w,lr_b,
  //   tok_idx,ln_w,ln_b,W1,b1,pn_w,pn_b}, 19..35 params_bwd, 36 gf, 37 gb,
  // 38..47 fep {fc1_w,fc1_b,fc2_w,fc2_b,g_w,g_b,dw1_w,dw1_b,dw2_w,dw2_b}
  const float* x = (const float*)d_in[0];

  char* p = (char*)d_ws;
  auto take = [&](size_t bytes)->char*{ char* r = p; p += (bytes + 255) & ~(size_t)255; return r; };
  float*  bufA  = (float*) take((size_t)MC*4);
  float*  bufB  = (float*) take((size_t)MC*4);
  float*  bufC  = (float*) take((size_t)MC*4);
  float*  bufOF = (float*) take((size_t)MC*4);
  float*  bufOB = (float*) take((size_t)MC*4);
  bf16_t* xn16  = (bf16_t*)take((size_t)MC*2);
  bf16_t* xf16  = (bf16_t*)take((size_t)MC*2);
  bf16_t* q16   = (bf16_t*)take((size_t)MC*2);
  bf16_t* k16   = (bf16_t*)take((size_t)MC*2);
  bf16_t* z16   = (bf16_t*)take((size_t)MC*2);
  bf16_t* w16   = (bf16_t*)take((size_t)11*CDIM*CDIM*2);
  bf16_t* w1t16 = (bf16_t*)take((size_t)2*NHEADS*64*64*2);
  float*  lre   = (float*) take((size_t)NPATCH*NHEADS*KLEN*4);

  const int WN = CDIM*CDIM;
  int ewg = (MC + 255)/256;
  dim3 ggrid(MROWS/128, CDIM/128);

  k_convert_x<<<ewg,256,0,stream>>>(x, xn16, xf16);
  int wgrid = (WN+255)/256;
  // slots: dir*4 + {wq,wv,wg,wo}; 8=fc1, 9=fc2, 10=g_w
  k_convert_plain<<<wgrid,256,0,stream>>>((const float*)d_in[2],  w16+(size_t)0*WN, WN);
  k_convert_plain<<<wgrid,256,0,stream>>>((const float*)d_in[3],  w16+(size_t)1*WN, WN);
  k_convert_plain<<<wgrid,256,0,stream>>>((const float*)d_in[5],  w16+(size_t)2*WN, WN);
  k_convert_plain<<<wgrid,256,0,stream>>>((const float*)d_in[4],  w16+(size_t)3*WN, WN);
  k_convert_plain<<<wgrid,256,0,stream>>>((const float*)d_in[19], w16+(size_t)4*WN, WN);
  k_convert_plain<<<wgrid,256,0,stream>>>((const float*)d_in[20], w16+(size_t)5*WN, WN);
  k_convert_plain<<<wgrid,256,0,stream>>>((const float*)d_in[22], w16+(size_t)6*WN, WN);
  k_convert_plain<<<wgrid,256,0,stream>>>((const float*)d_in[21], w16+(size_t)7*WN, WN);
  k_convert_plain<<<wgrid,256,0,stream>>>((const float*)d_in[38], w16+(size_t)8*WN, WN);
  k_convert_plain<<<wgrid,256,0,stream>>>((const float*)d_in[40], w16+(size_t)9*WN, WN);
  k_convert_plain<<<wgrid,256,0,stream>>>((const float*)d_in[42], w16+(size_t)10*WN, WN);
  int w1grid = (NHEADS*4096+255)/256;
  k_convert_w1t<<<w1grid,256,0,stream>>>((const float*)d_in[15], w1t16);
  k_convert_w1t<<<w1grid,256,0,stream>>>((const float*)d_in[32], w1t16 + NHEADS*4096);

  for (int dir = 0; dir < 2; ++dir){
    int P = dir ? 19 : 2;
    const bf16_t* xd = dir ? xf16 : xn16;
    float* odir = dir ? bufOB : bufOF;
    int ws0 = dir*4;
    // xq = x @ wq
    gemm_bf16_384<<<ggrid,256,0,stream>>>(xd, w16 + (size_t)(ws0+0)*WN, nullptr, bufA);
    // conv + rope -> XQ/XK (bf16 + f32 heads layout)
    k_conv_rope<<<(MROWS*(CDIM/2)+255)/256,256,0,stream>>>(bufA,
        (const float*)d_in[P+4], (const float*)d_in[P+5],
        (const float*)d_in[P+6], (const float*)d_in[P+7],
        q16, k16, bufC, bufB);
    // XV = x @ wv ; tgt = XV - XK (in place over bufB)
    gemm_bf16_384<<<ggrid,256,0,stream>>>(xd, w16 + (size_t)(ws0+1)*WN, nullptr, bufA);
    k_make_tgt<<<ewg,256,0,stream>>>(bufA, bufB);
    k_lr<<<(NPATCH*NHEADS*KLEN+255)/256,256,0,stream>>>(x,
        (const float*)d_in[P+8], (const float*)d_in[P+9], lre, dir);
    // fused dual-form inner step -> out_core [B,K,C] in bufA
    ttt_inner<<<dim3(NHEADS, NPATCH),256,0,stream>>>(q16, k16, bufC, bufB, lre,
        w1t16 + (size_t)dir*NHEADS*4096,
        (const float*)d_in[P+14], (const float*)d_in[P+11],
        (const float*)d_in[P+12], (const float*)d_in[P+10], bufA);
    // gate: gg = x @ wg ; z = LN(gelu(gg)*core) ; odir = z @ wo
    gemm_bf16_384<<<ggrid,256,0,stream>>>(xd, w16 + (size_t)(ws0+2)*WN, nullptr, bufC);
    k_gate_postnorm<<<MROWS,128,0,stream>>>(bufC, bufA,
        (const float*)d_in[P+15], (const float*)d_in[P+16], z16);
    gemm_bf16_384<<<ggrid,256,0,stream>>>(z16, w16 + (size_t)(ws0+3)*WN, nullptr, odir);
  }

  // fused = tanh(gf)*of + tanh(gb)*flip(ob) + x  -> y in bufA
  k_fuse<<<ewg,256,0,stream>>>(bufOF, bufOB, x, (const float*)d_in[36], (const float*)d_in[37], bufA);
  // FEP: h = dwconv3(y) ; u = silu(h@fc1+b)*(h@fc2+b) ; v = u@g_w+g_b ; out = y + dwconv3(v)
  k_dwconv3_bf16<<<ewg,256,0,stream>>>(bufA, (const float*)d_in[44], (const float*)d_in[45], z16);
  gemm_bf16_384<<<ggrid,256,0,stream>>>(z16, w16 + (size_t)8*WN, (const float*)d_in[39], bufB);
  gemm_bf16_384<<<ggrid,256,0,stream>>>(z16, w16 + (size_t)9*WN, (const float*)d_in[41], bufC);
  k_silumul<<<ewg,256,0,stream>>>(bufB, bufC, q16);
  gemm_bf16_384<<<ggrid,256,0,stream>>>(q16, w16 + (size_t)10*WN, (const float*)d_in[43], bufB);
  k_final<<<ewg,256,0,stream>>>(bufA, bufB, (const float*)d_in[46], (const float*)d_in[47], (float*)d_out);
}
